// SemEmbedding_68255620268077
// MI455X (gfx1250) — matrix operand order repaired and checked by reading, not compiled, not run on hardware
//
#include <hip/hip_runtime.h>
#include <hip/hip_bf16.h>
#include <stdint.h>

typedef __attribute__((ext_vector_type(16))) __bf16 v16bf;
typedef __attribute__((ext_vector_type(8)))  float  v8f;

#define B_    16
#define S_    1024
#define L_    512
#define D_    1024
#define H_    8
#define DH_   128
#define WD_   256

// ---------------------------------------------------------------------------
// bf16-WMMA GEMM, register-blocked 64x32 per wave (4 M-subtiles x 2 N-subtiles,
// 8 accumulators). Per K-step: 2 resident B-fragments; A-fragments are loaded
// one at a time and consumed immediately (short live range -> no spills).
//   out[m,n] = (sum_k A[m,k]*W[n,k] + bias[n]) * scale   [+ out[m,n] if ACCUM]
// A: [M,K] f32 row-major (lda), W: [N,K] f32 row-major (ldw).
// ---------------------------------------------------------------------------
template<bool HAS_BIAS, bool ACCUM>
__global__ __launch_bounds__(256, 1)
void gemm_wmma_bf16(const float* __restrict__ A, int lda,
                    const float* __restrict__ W, int ldw,
                    const float* __restrict__ bias,
                    float* __restrict__ out, int ldo,
                    int M, int N, int Kdim, float scale)
{
    const int lane = threadIdx.x & 31;
    const int wid  = threadIdx.x >> 5;
    const int w    = blockIdx.x * 8 + wid;
    const int ntn  = N >> 5;                    // 32-col strips
    const int tmb  = w / ntn;
    const int tnb  = w - tmb * ntn;
    if (tmb >= (M >> 6)) return;                // wave-uniform: EXEC stays full

    const int lo   = lane & 15;
    const int hi8  = (lane >> 4) * 8;
    const int hi16 = (lane >> 4) * 16;

    const float* arow[4];
#pragma unroll
    for (int i = 0; i < 4; ++i)
        arow[i] = A + (size_t)(tmb * 64 + i * 16 + lo) * (size_t)lda;
    const float* wrow[2];
#pragma unroll
    for (int j = 0; j < 2; ++j)
        wrow[j] = W + (size_t)(tnb * 32 + j * 16 + lo) * (size_t)ldw;

    v8f acc[4][2];
#pragma unroll
    for (int i = 0; i < 4; ++i)
#pragma unroll
        for (int j = 0; j < 2; ++j)
#pragma unroll
            for (int r = 0; r < 8; ++r) acc[i][j][r] = 0.0f;

    for (int k0 = 0; k0 < Kdim; k0 += 32) {
        v16bf bfm[2];
#pragma unroll
        for (int j = 0; j < 2; ++j)
#pragma unroll
            for (int e = 0; e < 16; ++e)
                bfm[j][e] = (__bf16)wrow[j][k0 + hi8 + (e & 7) + ((e >> 3) << 4)];  // K = hi8+e / 16+hi8+(e-8)

#pragma unroll
        for (int i = 0; i < 4; ++i) {
            v16bf af;
#pragma unroll
            for (int e = 0; e < 8; ++e) {
                af[e]     = (__bf16)arow[i][k0 + hi8 + e];           // K = hi8+e
                af[e + 8] = (__bf16)arow[i][k0 + 16 + hi8 + e];      // K = 16+hi8+e
            }
            acc[i][0] = __builtin_amdgcn_wmma_f32_16x16x32_bf16(
                false, af, false, bfm[0], (short)0, acc[i][0], false, false);
            acc[i][1] = __builtin_amdgcn_wmma_f32_16x16x32_bf16(
                false, af, false, bfm[1], (short)0, acc[i][1], false, false);
        }
    }

#pragma unroll
    for (int j = 0; j < 2; ++j) {
        const int n = tnb * 32 + j * 16 + lo;
        const float bv = HAS_BIAS ? bias[n] : 0.0f;
#pragma unroll
        for (int i = 0; i < 4; ++i)
#pragma unroll
            for (int r = 0; r < 8; ++r) {
                const int m = tmb * 64 + i * 16 + r + hi8;
                const size_t off = (size_t)m * (size_t)ldo + n;
                float v = (acc[i][j][r] + bv) * scale;
                if (ACCUM) v += out[off];
                out[off] = v;
            }
    }
}

// ---------------------------------------------------------------------------
// Fused masked flash-attention. One wave per (b, h, 16-row L tile); the 4
// waves of a block share (b,h), so the V chunk is staged block-wide into LDS
// with async global->LDS copies, double-buffered and overlapped with compute.
// ---------------------------------------------------------------------------
__global__ __launch_bounds__(128, 1)
void attn_wmma_kernel(const float* __restrict__ Q,
                      const float* __restrict__ K,
                      const float* __restrict__ V,
                      const int*   __restrict__ to_sem,
                      const int*   __restrict__ sem_syn,
                      float* __restrict__ ctx)
{
    __shared__ float  vbuf[2][32][132];         // 132-f32 rows: 16B-aligned + bank stagger
    __shared__ __bf16 plds[4][16][40];          // per-wave 16x32 P tile (padded)

    const int tid   = threadIdx.x;
    const int lane  = tid & 31;
    const int wid   = tid >> 5;
    const int gwave = blockIdx.x * 4 + wid;     // 0 .. B*H*(L/16)-1
    const int ltile = gwave & 31;               // L/16 == 32
    const int bh    = gwave >> 5;               // block-uniform
    const int h     = bh & 7;
    const int b     = bh >> 3;
    const int l0    = ltile * 16;

    const int lo   = lane & 15;
    const int hi8  = (lane >> 4) * 8;
    const int hi16 = (lane >> 4) * 16;

    const float* kbase = K + (size_t)b * S_ * D_ + h * DH_;
    const float* vbase = V + (size_t)b * S_ * D_ + h * DH_;
    const int*   ssb   = sem_syn + b * S_;

    // async-stage one 32x128 f32 V chunk (1024 x b128 pieces / 128 threads)
    auto prefetchV = [&](int s0, int buf) {
#pragma unroll
        for (int i = 0; i < 8; ++i) {
            const int p   = tid + i * 128;      // 0..1023
            const int row = p >> 5;             // 0..31
            const int c4  = (p & 31) * 4;       // 0..124
            const float* ga = vbase + (size_t)(s0 + row) * D_ + c4;
            const uint32_t loff = (uint32_t)(uintptr_t)&vbuf[buf][row][c4];
            asm volatile("global_load_async_to_lds_b128 %0, %1, off"
                         :: "v"(loff), "v"(ga) : "memory");
        }
    };

    // per-lane row metadata (rows r+hi8 of this tile)
    int tosem_row[8];
#pragma unroll
    for (int r = 0; r < 8; ++r)
        tosem_row[r] = to_sem[b * L_ + l0 + r + hi8];

    // preload Q A-fragments for the whole 16x128 row tile (4 K-chunks of 32)
    const float* qrow = Q + ((size_t)b * L_ + l0 + lo) * D_ + h * DH_;
    v16bf qfrag[4];
#pragma unroll
    for (int kk = 0; kk < 4; ++kk) {
#pragma unroll
        for (int e = 0; e < 8; ++e) {
            qfrag[kk][e]     = (__bf16)qrow[kk * 32 + hi8 + e];
            qfrag[kk][e + 8] = (__bf16)qrow[kk * 32 + 16 + hi8 + e];
        }
    }

    v8f ctxacc[8];
#pragma unroll
    for (int d = 0; d < 8; ++d)
#pragma unroll
        for (int i = 0; i < 8; ++i) ctxacc[d][i] = 0.0f;

    float rmax[8], rsum[8];
#pragma unroll
    for (int r = 0; r < 8; ++r) { rmax[r] = -3.0e38f; rsum[r] = 0.0f; }

    prefetchV(0, 0);

    for (int c = 0; c < S_ / 32; ++c) {
        const int s0  = c * 32;
        const int buf = c & 1;

        asm volatile("s_wait_asynccnt 0x0" ::: "memory");
        __syncthreads();                        // V chunk `buf` visible to all waves
        if (c + 1 < S_ / 32) prefetchV(s0 + 32, buf ^ 1);

        // ---- scores: two 16x16 tiles covering cols s0..s0+31 ----
        v8f sc0, sc1;
#pragma unroll
        for (int i = 0; i < 8; ++i) { sc0[i] = 0.0f; sc1[i] = 0.0f; }

        const float* krow0 = kbase + (size_t)(s0 + lo) * D_;
        const float* krow1 = kbase + (size_t)(s0 + 16 + lo) * D_;
#pragma unroll
        for (int kk = 0; kk < 4; ++kk) {
            v16bf bf0, bf1;
#pragma unroll
            for (int e = 0; e < 16; ++e) {
                bf0[e] = (__bf16)krow0[kk * 32 + hi8 + (e & 7) + ((e >> 3) << 4)];
                bf1[e] = (__bf16)krow1[kk * 32 + hi8 + (e & 7) + ((e >> 3) << 4)];
            }
            sc0 = __builtin_amdgcn_wmma_f32_16x16x32_bf16(false, qfrag[kk], false, bf0,
                                                          (short)0, sc0, false, false);
            sc1 = __builtin_amdgcn_wmma_f32_16x16x32_bf16(false, qfrag[kk], false, bf1,
                                                          (short)0, sc1, false, false);
        }

        // ---- mask + online softmax (rows r+hi8 live in this lane half) ----
        const int scol0 = s0 + lo;
        const int scol1 = s0 + 16 + lo;
        const int ss0 = ssb[scol0];
        const int ss1 = ssb[scol1];

        float p0[8], p1[8], fac[8];
#pragma unroll
        for (int r = 0; r < 8; ++r) {
            const int lrow = l0 + r + hi8;
            float v0 = sc0[r], v1 = sc1[r];
            const bool a0 = (ss0 == lrow + 1) || (scol0 == tosem_row[r]);
            const bool a1 = (ss1 == lrow + 1) || (scol1 == tosem_row[r]);
            v0 = a0 ? v0 : -1.0e30f;
            v1 = a1 ? v1 : -1.0e30f;
            float m = fmaxf(v0, v1);
#pragma unroll
            for (int o = 8; o >= 1; o >>= 1) m = fmaxf(m, __shfl_xor(m, o, 16));
            const float mnew = fmaxf(rmax[r], m);
            const float f = __expf(rmax[r] - mnew);
            rmax[r] = mnew;
            const float e0 = __expf(v0 - mnew);
            const float e1 = __expf(v1 - mnew);
            float psum = e0 + e1;
#pragma unroll
            for (int o = 8; o >= 1; o >>= 1) psum += __shfl_xor(psum, o, 16);
            rsum[r] = rsum[r] * f + psum;
            fac[r] = f; p0[r] = e0; p1[r] = e1;
        }

        // rescale running ctx (same C-layout row mapping as stats)
#pragma unroll
        for (int d = 0; d < 8; ++d)
#pragma unroll
            for (int r = 0; r < 8; ++r) ctxacc[d][r] *= fac[r];

        // ---- relayout P through LDS into A-fragment form ----
#pragma unroll
        for (int r = 0; r < 8; ++r) {
            plds[wid][r + hi8][lo]      = (__bf16)p0[r];
            plds[wid][r + hi8][16 + lo] = (__bf16)p1[r];
        }
        __syncthreads();

        v16bf pfrag;
#pragma unroll
        for (int e = 0; e < 8; ++e) {
            pfrag[e]     = plds[wid][lo][hi8 + e];
            pfrag[e + 8] = plds[wid][lo][16 + hi8 + e];
        }
        __syncthreads();

        // ---- ctx += P(16x32) @ V(32x128), V from LDS stage ----
#pragma unroll
        for (int d = 0; d < 8; ++d) {
            v16bf vfrag;
#pragma unroll
            for (int e = 0; e < 16; ++e)
                vfrag[e] = (__bf16)vbuf[buf][hi8 + (e & 7) + ((e >> 3) << 4)][d * 16 + lo];
            ctxacc[d] = __builtin_amdgcn_wmma_f32_16x16x32_bf16(false, pfrag, false, vfrag,
                                                                (short)0, ctxacc[d], false, false);
        }
        __syncthreads();                        // all waves done with `buf` before refill
    }

    // ---- normalize + write ----
#pragma unroll
    for (int d = 0; d < 8; ++d)
#pragma unroll
        for (int r = 0; r < 8; ++r) {
            const int lrow = l0 + r + hi8;
            ctx[((size_t)b * L_ + lrow) * D_ + h * DH_ + d * 16 + lo] =
                ctxacc[d][r] / rsum[r];
        }
}

// ---------------------------------------------------------------------------
// Gathers and output assembly
// ---------------------------------------------------------------------------
__global__ void gather_sem_kernel(const float* __restrict__ enc,
                                  const int* __restrict__ to_sem,
                                  float* __restrict__ dst)
{
    const int row = blockIdx.x;                 // b*L + l
    const int b = row / L_;
    const int s = to_sem[row];
    const float4* src = (const float4*)(enc + ((size_t)b * S_ + s) * D_);
    float4* d = (float4*)(dst + (size_t)row * D_);
    d[threadIdx.x] = src[threadIdx.x];          // 256 threads * 4 floats = 1024
}

__global__ void gather_type_kernel(const float* __restrict__ emb,
                                   const int* __restrict__ types,
                                   float* __restrict__ dst)
{
    const int row = blockIdx.x;                 // b*L + l
    const int t = types[row];
    dst[(size_t)row * WD_ + threadIdx.x] = emb[(size_t)t * WD_ + threadIdx.x];
}

__global__ void concat_kernel(const float* __restrict__ emb,
                              const float* __restrict__ root,
                              float* __restrict__ out2)
{
    const int row = blockIdx.x;                 // b*(L+1) + i
    const int b = row / (L_ + 1);
    const int i = row - b * (L_ + 1);
    float4* d = (float4*)(out2 + (size_t)row * D_);
    const float4* s = (i == 0)
        ? (const float4*)root
        : (const float4*)(emb + ((size_t)b * L_ + (i - 1)) * D_);
    d[threadIdx.x] = s[threadIdx.x];
}

// ---------------------------------------------------------------------------
extern "C" void kernel_launch(void* const* d_in, const int* in_sizes, int n_in,
                              void* d_out, int out_size, void* d_ws, size_t ws_size,
                              hipStream_t stream)
{
    const float* enc     = (const float*)d_in[0];
    const int*   to_sem  = (const int*)  d_in[1];
    const int*   to_type = (const int*)  d_in[2];
    const int*   sem_syn = (const int*)  d_in[3];
    const float* wq = (const float*)d_in[4];
    const float* bq = (const float*)d_in[5];
    const float* wk = (const float*)d_in[6];
    const float* bk = (const float*)d_in[7];
    const float* wv = (const float*)d_in[8];
    const float* bv = (const float*)d_in[9];
    const float* wo = (const float*)d_in[10];
    const float* bo = (const float*)d_in[11];
    const float* temb  = (const float*)d_in[12];
    const float* wproj = (const float*)d_in[13];
    const float* bproj = (const float*)d_in[14];
    const float* broot = (const float*)d_in[15];

    float* ws      = (float*)d_ws;
    float* buf_sem = ws;                                   // B*L*D (reused as ctx)
    float* buf_q   = buf_sem + (size_t)B_ * L_ * D_;       // B*L*D (reused as attn_out)
    float* buf_k   = buf_q   + (size_t)B_ * L_ * D_;       // B*S*D
    float* buf_v   = buf_k   + (size_t)B_ * S_ * D_;       // B*S*D
    float* buf_t   = buf_v   + (size_t)B_ * S_ * D_;       // B*L*WD

    float* out_emb = (float*)d_out;                        // B*L*D
    float* out_cat = out_emb + (size_t)B_ * L_ * D_;       // B*(L+1)*D

    // 1. gathers
    gather_sem_kernel <<<B_ * L_, 256, 0, stream>>>(enc, to_sem, buf_sem);
    gather_type_kernel<<<B_ * L_, 256, 0, stream>>>(temb, to_type, buf_t);

    const float qscale = 0.08838834764831845f;             // DH^-0.5

    // 2. Q/K/V projections (64x32 tiles per wave, 8 waves/block)
    {   const int M = B_ * L_, N = D_, tiles = (M / 64) * (N / 32);
        gemm_wmma_bf16<true, false><<<tiles / 8, 256, 0, stream>>>(
            buf_sem, D_, wq, D_, bq, buf_q, D_, M, N, D_, qscale); }
    {   const int M = B_ * S_, N = D_, tiles = (M / 64) * (N / 32);
        gemm_wmma_bf16<true, false><<<tiles / 8, 256, 0, stream>>>(
            enc, D_, wk, D_, bk, buf_k, D_, M, N, D_, 1.0f); }
    {   const int M = B_ * S_, N = D_, tiles = (M / 64) * (N / 32);
        gemm_wmma_bf16<true, false><<<tiles / 8, 256, 0, stream>>>(
            enc, D_, wv, D_, bv, buf_v, D_, M, N, D_, 1.0f); }

    // 3. fused masked attention -> ctx (reuses sem buffer)
    attn_wmma_kernel<<<(B_ * H_ * (L_ / 16)) / 4, 128, 0, stream>>>(
        buf_q, buf_k, buf_v, to_sem, sem_syn, buf_sem);

    // 4. out-projection -> attn_out (reuses Q buffer)
    {   const int M = B_ * L_, N = D_, tiles = (M / 64) * (N / 32);
        gemm_wmma_bf16<true, false><<<tiles / 8, 256, 0, stream>>>(
            buf_sem, D_, wo, D_, bo, buf_q, D_, M, N, D_, 1.0f); }

    // 5. concat projection split in two accumulating passes
    {   const int M = B_ * L_, N = D_, tiles = (M / 64) * (N / 32);
        gemm_wmma_bf16<true, false><<<tiles / 8, 256, 0, stream>>>(
            buf_q, D_, wproj, D_ + WD_, bproj, out_emb, D_, M, N, D_, 1.0f);
        gemm_wmma_bf16<false, true><<<tiles / 8, 256, 0, stream>>>(
            buf_t, WD_, wproj + D_, D_ + WD_, nullptr, out_emb, D_, M, N, WD_, 1.0f); }

    // 6. second output: root-prepended concat
    concat_kernel<<<B_ * (L_ + 1), 256, 0, stream>>>(out_emb, broot, out_cat);
}
